// FastKANLayer_11982958756320
// MI455X (gfx1250) — compile-verified, matrix operand order checked by reading, not run on hardware
//
#include <hip/hip_runtime.h>
#include <hip/hip_bf16.h>

// ---------------- problem constants ----------------
constexpr int B_  = 8;    // batch
constexpr int IN_ = 8;    // input channels
constexpr int J_  = 8;    // output channels ("out_n")
constexpr int S_  = 187;  // seq
constexpr int SP_ = 192;  // seq padded to 12*16 for WMMA tiling
constexpr int F_  = 64;   // feat
constexpr int O_  = 64;   // proj dim
constexpr int C_  = 67;   // spline coeffs
constexpr int KS_ = 15;   // depthwise kernel size
constexpr int E_  = 256;  // mid*d = 4*64 attention embed dim
constexpr int BJ_ = B_ * J_;

typedef __attribute__((ext_vector_type(16))) _Float16 v16h;
typedef __attribute__((ext_vector_type(8)))  _Float16 v8h;
typedef __attribute__((ext_vector_type(8)))  float    v8f;

union AB16 { v16h v; v8h h2[2]; _Float16 h[16]; };
union F8   { v8f  v; float f[8]; };

// ---------------- small helpers ----------------
__device__ __forceinline__ float blkSum256(float v, float* red) {
  int t = threadIdx.x;
  red[t] = v; __syncthreads();
  for (int st = 128; st > 0; st >>= 1) { if (t < st) red[t] += red[t + st]; __syncthreads(); }
  float r = red[0]; __syncthreads();
  return r;
}
__device__ __forceinline__ float blkMax256(float v, float* red) {
  int t = threadIdx.x;
  red[t] = v; __syncthreads();
  for (int st = 128; st > 0; st >>= 1) { if (t < st) red[t] = fmaxf(red[t], red[t + st]); __syncthreads(); }
  float r = red[0]; __syncthreads();
  return r;
}
__device__ __forceinline__ float blkSum64(float v, float* red) {
  int t = threadIdx.x;
  red[t] = v; __syncthreads();
  for (int st = 32; st > 0; st >>= 1) { if (t < st) red[t] += red[t + st]; __syncthreads(); }
  float r = red[0]; __syncthreads();
  return r;
}

// 4-tap cubic B-spline evaluation around normalized coordinate tt = (xn+1)/h
__device__ __forceinline__ void splineTaps(float tt, float* bs, int* ks) {
  int kf = (int)floorf(tt);
#pragma unroll
  for (int q = 0; q < 4; ++q) {
    int k = kf - 1 + q;
    ks[q] = k;
    float d  = fabsf(tt - (float)k);
    float r2 = fmaxf(2.0f - d, 0.0f);
    float r1 = fmaxf(1.0f - d, 0.0f);
    float bb = r2 * r2 * r2 * (1.0f / 6.0f) - r1 * r1 * r1 * (4.0f / 6.0f);
    bs[q] = (k >= 0 && k < C_) ? bb : 0.0f;
  }
}

// ---------------- kernel 1: global max|x| ----------------
__global__ void k_maxabs(const float* __restrict__ x, int n, unsigned int* out) {
  float m = 0.0f;
  for (int i = blockIdx.x * blockDim.x + threadIdx.x; i < n; i += gridDim.x * blockDim.x)
    m = fmaxf(m, fabsf(x[i]));
  __shared__ float red[256];
  m = blkMax256(m, red);
  if (threadIdx.x == 0) atomicMax(out, __float_as_uint(m));  // all values >= 0
}

// ---------------- kernel 2: energies[b,i,j] ----------------
__global__ void k_energies(const float* __restrict__ x_in, const float* __restrict__ sw,
                           const float* __restrict__ omiga, const unsigned int* __restrict__ mau,
                           float* __restrict__ energies) {
  int bi = blockIdx.x;             // b*IN + i
  int i  = bi & 7;
  __shared__ float s_sw[J_ * C_];
  __shared__ float s_om[J_];
  for (int t = threadIdx.x; t < J_ * C_; t += blockDim.x) s_sw[t] = sw[i * J_ * C_ + t];
  if (threadIdx.x < J_) s_om[threadIdx.x] = fabsf(omiga[i * J_ + threadIdx.x]);
  __syncthreads();
  float maxv = __uint_as_float(*mau) + 1e-8f;
  float inv_scale = 0.95f / maxv;
  float e[J_];
#pragma unroll
  for (int j = 0; j < J_; ++j) e[j] = 0.0f;
  for (int t = threadIdx.x; t < S_ * F_; t += blockDim.x) {
    float xv = x_in[(size_t)bi * S_ * F_ + t];
    float xn = fminf(0.99f, fmaxf(-0.99f, xv * inv_scale));
    float tt = (xn + 1.0f) * ((float)(C_ - 1) * 0.5f);   // 1/h = 33
    float bs[4]; int ks[4];
    splineTaps(tt, bs, ks);
#pragma unroll
    for (int j = 0; j < J_; ++j) {
      float sv = 0.0f;
#pragma unroll
      for (int q = 0; q < 4; ++q) {
        int k = ks[q];
        if (k >= 0 && k < C_) sv += bs[q] * s_sw[j * C_ + k];
      }
      float a = sv + s_om[j] * xv;
      e[j] += a * a;
    }
  }
  __shared__ float red[256];
#pragma unroll
  for (int j = 0; j < J_; ++j) {
    float s = blkSum256(e[j], red);
    if (threadIdx.x == 0) energies[bi * J_ + j] = s * (1.0f / (float)(S_ * F_));
  }
}

// ---------------- kernel 3: gating (+ precomputed score scale) ----------------
__global__ void k_gate(const float* __restrict__ energies, const float* __restrict__ tau,
                       const float* __restrict__ temperature,
                       float* __restrict__ mask, float* __restrict__ gmm, float* scale_out) {
  int idx = blockIdx.x * blockDim.x + threadIdx.x;
  float tv = fabsf(temperature[0]) * sqrtf((float)S_ / 256.0f) + 1e-4f;
  // raw scores are divided by sqrt(mid*d)*t_val = 16*t_val; store the reciprocal once
  if (idx == 0) *scale_out = 1.0f / (16.0f * tv);
  if (idx < B_ * IN_ * J_) {
    int ij = idx & 63;
    float ta = fabsf(tau[ij]);
    float sq = sqrtf(energies[idx] + 1e-8f);
    float m  = 1.0f / (1.0f + expf(-(sq - ta) / tv));
    mask[idx] = m;
    gmm[idx]  = m * (sq / (ta + 1e-8f));
  }
}

// ---------------- kernel 4: gated projections -> LN -> f16 Kt/Qt (padded) ----------------
__global__ void k_ktqt(const float* __restrict__ proj, const float* __restrict__ gmm,
                       _Float16* __restrict__ Kth, _Float16* __restrict__ Qth) {
  int blk = blockIdx.x;           // bj*SP + sp
  int bj  = blk / SP_, sp = blk % SP_;
  int b = bj >> 3, j = bj & 7;
  int e = threadIdx.x;            // 0..255
  _Float16* Krow = Kth + ((size_t)bj * SP_ + sp) * E_;
  _Float16* Qrow = Qth + ((size_t)bj * SP_ + sp) * E_;
  if (sp >= S_) { Krow[e] = (_Float16)0.0f; Qrow[e] = (_Float16)0.0f; return; }
  int ih = e >> 6, d = e & 63;
  int iK = 2 * ih, iQ = iK + 1;
  float kv = proj[(((size_t)b * IN_ + iK) * S_ + sp) * O_ + d] * gmm[(b * IN_ + iK) * J_ + j];
  float qv = proj[(((size_t)b * IN_ + iQ) * S_ + sp) * O_ + d] * gmm[(b * IN_ + iQ) * J_ + j];
  __shared__ float red[256];
  float ksum = blkSum256(kv, red)      * (1.0f / 256.0f);
  float ksq  = blkSum256(kv * kv, red) * (1.0f / 256.0f);
  float qsum = blkSum256(qv, red)      * (1.0f / 256.0f);
  float qsq  = blkSum256(qv * qv, red) * (1.0f / 256.0f);
  float kn = (kv - ksum) * rsqrtf(ksq - ksum * ksum + 1e-5f);
  float qn = (qv - qsum) * rsqrtf(qsq - qsum * qsum + 1e-5f);
  Krow[e] = (_Float16)kn;
  Qrow[e] = (_Float16)qn;
}

// ---------------- kernel 5: raw = Kt x Qt^T  (WMMA f16, K=256) ----------------
__global__ void k_scores(const _Float16* __restrict__ Kth, const _Float16* __restrict__ Qth,
                         const float* __restrict__ qscale, float* __restrict__ rawf) {
  int mt = blockIdx.x, nt = blockIdx.y, bj = blockIdx.z;
  int l = threadIdx.x, lo = l & 15, hi = l >> 4;
  const _Float16* Arow = Kth + ((size_t)bj * SP_ + mt * 16 + lo) * E_;
  const _Float16* Brow = Qth + ((size_t)bj * SP_ + nt * 16 + lo) * E_;
  F8 acc; acc.v = v8f{};
#pragma unroll
  for (int kk = 0; kk < E_; kk += 32) {
    AB16 a, bm;
    a.h2[0]  = *(const v8h*)(Arow + kk + hi * 8);
    a.h2[1]  = *(const v8h*)(Arow + kk + hi * 8 + 16);
    bm.h2[0] = *(const v8h*)(Brow + kk + hi * 8);
    bm.h2[1] = *(const v8h*)(Brow + kk + hi * 8 + 16);
    acc.v = __builtin_amdgcn_wmma_f32_16x16x32_f16(false, a.v, false, bm.v,
                                                   (short)0, acc.v, false, false);
  }
  float scale = qscale[0];          // precomputed reciprocal: no per-block divide
  float* out = rawf + (size_t)bj * SP_ * SP_;
#pragma unroll
  for (int r = 0; r < 8; ++r) {
    int m = mt * 16 + r + hi * 8;
    int n = nt * 16 + lo;
    out[(size_t)m * SP_ + n] = acc.f[r] * scale;
  }
}

// ---------------- kernel 6: softmax over t (<S), write f16 padded W1 ----------------
__global__ void k_softmax(const float* __restrict__ rawf, _Float16* __restrict__ W1h) {
  int bj = blockIdx.x / SP_, sp = blockIdx.x % SP_;
  _Float16* orow = W1h + ((size_t)bj * SP_ + sp) * SP_;
  int tid = threadIdx.x;  // 256
  if (sp >= S_) { if (tid < SP_) orow[tid] = (_Float16)0.0f; return; }
  const float* irow = rawf + ((size_t)bj * SP_ + sp) * SP_;
  float v = (tid < S_) ? irow[tid] : -3.0e38f;
  __shared__ float red[256];
  float mx = blkMax256(v, red);
  float ex = (tid < S_) ? expf(v - mx) : 0.0f;
  float sm = blkSum256(ex, red);
  float inv = 1.0f / sm;
  if (tid < SP_) orow[tid] = (_Float16)((tid < S_) ? ex * inv : 0.0f);
}

// ---------------- kernel 7: combined + LN -> f16 (padded); recomputes spline ----------------
__global__ void k_combined(const float* __restrict__ x_in, const float* __restrict__ sw,
                           const float* __restrict__ omiga, const float* __restrict__ mask,
                           const unsigned int* __restrict__ mau,
                           const float* __restrict__ ln_scale, const float* __restrict__ ln_bias,
                           float* __restrict__ combined, _Float16* __restrict__ xlnh) {
  int bj = blockIdx.x / SP_, sp = blockIdx.x % SP_;
  int b = bj >> 3, j = bj & 7;
  int f = threadIdx.x;  // 64
  _Float16* xrow = xlnh + ((size_t)bj * SP_ + sp) * F_;
  if (sp >= S_) { xrow[f] = (_Float16)0.0f; return; }
  __shared__ float s_sw[IN_ * C_];
  __shared__ float s_m[IN_];
  __shared__ float s_om[IN_];
  for (int t = f; t < IN_ * C_; t += 64) {
    int i = t / C_, k = t % C_;
    s_sw[t] = sw[(i * J_ + j) * C_ + k];
  }
  if (f < IN_) {
    s_m[f]  = mask[(b * IN_ + f) * J_ + j];
    s_om[f] = fabsf(omiga[f * J_ + j]);
  }
  __syncthreads();
  float maxv = __uint_as_float(*mau) + 1e-8f;
  float inv_scale = 0.95f / maxv;
  float acc = 0.0f;
#pragma unroll
  for (int i = 0; i < IN_; ++i) {
    float xv = x_in[(((size_t)b * IN_ + i) * S_ + sp) * F_ + f];
    float xn = fminf(0.99f, fmaxf(-0.99f, xv * inv_scale));
    float tt = (xn + 1.0f) * ((float)(C_ - 1) * 0.5f);
    float bs[4]; int ks[4];
    splineTaps(tt, bs, ks);
    float sv = 0.0f;
#pragma unroll
    for (int q = 0; q < 4; ++q) {
      int k = ks[q];
      if (k >= 0 && k < C_) sv += bs[q] * s_sw[i * C_ + k];
    }
    acc += s_m[i] * (sv + s_om[i] * xv);
  }
  combined[((size_t)bj * S_ + sp) * F_ + f] = acc;
  __shared__ float red[64];
  float m  = blkSum64(acc, red)       * (1.0f / 64.0f);
  float sq = blkSum64(acc * acc, red) * (1.0f / 64.0f);
  float xn = (acc - m) * rsqrtf(sq - m * m + 1e-5f);
  xrow[f] = (_Float16)(xn * ln_scale[j * F_ + f] + ln_bias[j * F_ + f]);
}

// ---------------- kernel 8: W2 -> f16 transposed [j][o][h] ----------------
__global__ void k_w2cvt(const float* __restrict__ W2, _Float16* __restrict__ W2t) {
  int idx = blockIdx.x * 256 + threadIdx.x;
  if (idx >= J_ * F_ * O_) return;
  int j = idx / (F_ * O_);
  int r = idx - j * F_ * O_;
  int h = r / O_, o = r % O_;
  W2t[((size_t)j * O_ + o) * F_ + h] = (_Float16)W2[idx];
}

// ---------------- kernel 9: x_prime = xln x W2 + b  (WMMA, K=64) ----------------
__global__ void k_xprime(const _Float16* __restrict__ xlnh, const _Float16* __restrict__ W2t,
                         const float* __restrict__ bp, float* __restrict__ xprime_out,
                         _Float16* __restrict__ xpT) {
  int mt = blockIdx.x, nt = blockIdx.y, bj = blockIdx.z;
  int j = bj & 7;
  int l = threadIdx.x, lo = l & 15, hi = l >> 4;
  const _Float16* Arow = xlnh + ((size_t)bj * SP_ + mt * 16 + lo) * F_;
  const _Float16* Brow = W2t  + ((size_t)j * O_ + nt * 16 + lo) * F_;
  F8 acc; acc.v = v8f{};
#pragma unroll
  for (int kk = 0; kk < F_; kk += 32) {
    AB16 a, bm;
    a.h2[0]  = *(const v8h*)(Arow + kk + hi * 8);
    a.h2[1]  = *(const v8h*)(Arow + kk + hi * 8 + 16);
    bm.h2[0] = *(const v8h*)(Brow + kk + hi * 8);
    bm.h2[1] = *(const v8h*)(Brow + kk + hi * 8 + 16);
    acc.v = __builtin_amdgcn_wmma_f32_16x16x32_f16(false, a.v, false, bm.v,
                                                   (short)0, acc.v, false, false);
  }
#pragma unroll
  for (int r = 0; r < 8; ++r) {
    int s = mt * 16 + r + hi * 8;
    int o = nt * 16 + lo;
    if (s < S_) {
      float v = acc.f[r] + bp[((size_t)j * S_ + s) * O_ + o];
      xprime_out[((size_t)bj * S_ + s) * O_ + o] = v;            // fp32 second output
      xpT[((size_t)bj * O_ + o) * SP_ + s] = (_Float16)v;        // transposed f16 for attn GEMM
    } else {
      xpT[((size_t)bj * O_ + o) * SP_ + s] = (_Float16)0.0f;
    }
  }
}

// ---------------- kernel 10: gp = W1 x x_prime (WMMA, K=192) ----------------
__global__ void k_gp(const _Float16* __restrict__ W1h, const _Float16* __restrict__ xpT,
                     float* __restrict__ gp) {
  int mt = blockIdx.x, nt = blockIdx.y, bj = blockIdx.z;
  int l = threadIdx.x, lo = l & 15, hi = l >> 4;
  const _Float16* Arow = W1h + ((size_t)bj * SP_ + mt * 16 + lo) * SP_;
  const _Float16* Brow = xpT + ((size_t)bj * O_ + nt * 16 + lo) * SP_;
  F8 acc; acc.v = v8f{};
#pragma unroll
  for (int kk = 0; kk < SP_; kk += 32) {
    AB16 a, bm;
    a.h2[0]  = *(const v8h*)(Arow + kk + hi * 8);
    a.h2[1]  = *(const v8h*)(Arow + kk + hi * 8 + 16);
    bm.h2[0] = *(const v8h*)(Brow + kk + hi * 8);
    bm.h2[1] = *(const v8h*)(Brow + kk + hi * 8 + 16);
    acc.v = __builtin_amdgcn_wmma_f32_16x16x32_f16(false, a.v, false, bm.v,
                                                   (short)0, acc.v, false, false);
  }
#pragma unroll
  for (int r = 0; r < 8; ++r) {
    int s = mt * 16 + r + hi * 8;
    int o = nt * 16 + lo;
    if (s < S_) gp[((size_t)bj * S_ + s) * O_ + o] = acc.f[r];
  }
}

// ---------------- kernel 11: depthwise conv + final fuse ----------------
__global__ void k_final(const float* __restrict__ gp, const float* __restrict__ xprime,
                        const float* __restrict__ combined, const float* __restrict__ alpha,
                        const float* __restrict__ beta, const float* __restrict__ theta,
                        const float* __restrict__ gamma, const float* __restrict__ w3,
                        float* __restrict__ out) {
  int idx = blockIdx.x * 256 + threadIdx.x;
  if (idx >= BJ_ * S_ * O_) return;
  int o = idx & 63;
  int rest = idx >> 6;
  int s = rest % S_;
  int bj = rest / S_;
  int j = bj & 7;
  const float* xp = xprime + (size_t)bj * S_ * O_;
  float wconv = 0.0f;
#pragma unroll
  for (int k = 0; k < KS_; ++k) {
    int t = s + k - KS_ / 2;
    if (t >= 0 && t < S_) wconv += xp[(size_t)t * O_ + o] * w3[(j * O_ + o) * KS_ + k];
  }
  float xv = xp[(size_t)s * O_ + o];
  out[idx] = fabsf(beta[j]) * gp[idx] + fabsf(alpha[j]) * xv +
             fabsf(theta[j]) * wconv + gamma[j] * combined[idx];
}

// ---------------- launch ----------------
extern "C" void kernel_launch(void* const* d_in, const int* in_sizes, int n_in,
                              void* d_out, int out_size, void* d_ws, size_t ws_size,
                              hipStream_t stream) {
  const float* x_in  = (const float*)d_in[0];
  const float* proj  = (const float*)d_in[1];
  const float* sw    = (const float*)d_in[2];
  const float* tau   = (const float*)d_in[3];
  const float* temp  = (const float*)d_in[4];
  const float* omiga = (const float*)d_in[5];
  const float* W2    = (const float*)d_in[6];
  const float* bp    = (const float*)d_in[7];
  const float* lns   = (const float*)d_in[8];
  const float* lnb   = (const float*)d_in[9];
  const float* alpha = (const float*)d_in[10];
  const float* beta  = (const float*)d_in[11];
  const float* theta = (const float*)d_in[12];
  const float* gamma = (const float*)d_in[13];
  const float* w3    = (const float*)d_in[14];

  float* out_main = (float*)d_out;                               // [B,J,S,O]
  float* xprime   = (float*)d_out + (size_t)BJ_ * S_ * O_;       // [B,J,S,O] (2nd output)

  // workspace carve (256B aligned)
  char* w = (char*)d_ws;
  size_t cur = 0;
  auto alloc = [&](size_t bytes) -> size_t {
    size_t off = cur;
    cur += (bytes + 255) & ~(size_t)255;
    return off;
  };
  unsigned int* maxabs = (unsigned int*)(w + alloc(4));
  float* qscale   = (float*)(w + alloc(4));
  float* energies = (float*)(w + alloc((size_t)B_ * IN_ * J_ * 4));
  float* mask     = (float*)(w + alloc((size_t)B_ * IN_ * J_ * 4));
  float* gmm      = (float*)(w + alloc((size_t)B_ * IN_ * J_ * 4));
  _Float16* Kth   = (_Float16*)(w + alloc((size_t)BJ_ * SP_ * E_ * 2));
  _Float16* Qth   = (_Float16*)(w + alloc((size_t)BJ_ * SP_ * E_ * 2));
  float* rawf     = (float*)(w + alloc((size_t)BJ_ * SP_ * SP_ * 4));
  _Float16* W1h   = (_Float16*)(w + alloc((size_t)BJ_ * SP_ * SP_ * 2));
  float* combined = (float*)(w + alloc((size_t)BJ_ * S_ * F_ * 4));
  _Float16* xlnh  = (_Float16*)(w + alloc((size_t)BJ_ * SP_ * F_ * 2));
  _Float16* W2t   = (_Float16*)(w + alloc((size_t)J_ * F_ * O_ * 2));
  _Float16* xpT   = (_Float16*)(w + alloc((size_t)BJ_ * O_ * SP_ * 2));
  float* gp       = (float*)(w + alloc((size_t)BJ_ * S_ * O_ * 4));
  (void)ws_size; (void)in_sizes; (void)n_in; (void)out_size;

  const int NX = B_ * IN_ * S_ * F_;

  hipMemsetAsync(maxabs, 0, sizeof(unsigned int), stream);
  k_maxabs<<<512, 256, 0, stream>>>(x_in, NX, maxabs);
  k_energies<<<B_ * IN_, 256, 0, stream>>>(x_in, sw, omiga, maxabs, energies);
  k_gate<<<2, 256, 0, stream>>>(energies, tau, temp, mask, gmm, qscale);
  k_ktqt<<<BJ_ * SP_, 256, 0, stream>>>(proj, gmm, Kth, Qth);
  k_scores<<<dim3(SP_ / 16, SP_ / 16, BJ_), 32, 0, stream>>>(Kth, Qth, qscale, rawf);
  k_softmax<<<BJ_ * SP_, 256, 0, stream>>>(rawf, W1h);
  k_w2cvt<<<(J_ * F_ * O_ + 255) / 256, 256, 0, stream>>>(W2, W2t);
  k_combined<<<BJ_ * SP_, 64, 0, stream>>>(x_in, sw, omiga, mask, maxabs, lns, lnb,
                                           combined, xlnh);
  k_xprime<<<dim3(SP_ / 16, O_ / 16, BJ_), 32, 0, stream>>>(xlnh, W2t, bp, xprime, xpT);
  k_gp<<<dim3(SP_ / 16, O_ / 16, BJ_), 32, 0, stream>>>(W1h, xpT, gp);
  k_final<<<(BJ_ * S_ * O_ + 255) / 256, 256, 0, stream>>>(gp, xprime, combined,
                                                           alpha, beta, theta, gamma, w3, out_main);
}
